// TemporalAdjLearner_21320217658126
// MI455X (gfx1250) — compile-verified, matrix operand order checked by reading, not run on hardware
//
#include <hip/hip_runtime.h>
#include <math.h>

#define N_ROWS 16384
#define T_DIM 8
#define D_DIM 128
#define KD_DIM 64
#define TOPK 8
#define CHUNK 1024          // columns per LDS chunk -> 16*1024*4 = 64 KB LDS
#define ROWS_PER_BLK 16

typedef __bf16 bf16_t;
typedef __attribute__((ext_vector_type(8)))  __bf16 v8bf;
typedef __attribute__((ext_vector_type(16))) __bf16 v16bf;
typedef __attribute__((ext_vector_type(8)))  float  v8f;

// ---------------------------------------------------------------------------
// Kernel A: temporal mean pool + Q/K projection, emit bf16 Q (pre-scaled by
// 1/sqrt(KD)=0.125) and bf16 K, row-major [N, 64].
// One block per row, 64 threads (2 waves); pooled row cached in LDS.
// ---------------------------------------------------------------------------
__global__ __launch_bounds__(64) void pool_project_kernel(
    const float* __restrict__ U,  const float* __restrict__ Wq,
    const float* __restrict__ bq, const float* __restrict__ Wk,
    const float* __restrict__ bk,
    bf16_t* __restrict__ Qb, bf16_t* __restrict__ Kb)
{
    __shared__ float pool[D_DIM];
    const int row = blockIdx.x;
    const int j   = threadIdx.x;            // 0..63
    const float* Urow = U + (size_t)row * (T_DIM * D_DIM);

    float s0 = 0.f, s1 = 0.f;
    #pragma unroll
    for (int t = 0; t < T_DIM; ++t) {
        s0 += Urow[t * D_DIM + j];
        s1 += Urow[t * D_DIM + j + 64];
    }
    pool[j]      = s0 * (1.0f / T_DIM);
    pool[j + 64] = s1 * (1.0f / T_DIM);
    __syncthreads();

    float q = bq[j], k = bk[j];
    #pragma unroll 4
    for (int d = 0; d < D_DIM; ++d) {
        const float p = pool[d];
        q = fmaf(p, Wq[d * KD_DIM + j], q);
        k = fmaf(p, Wk[d * KD_DIM + j], k);
    }
    Qb[(size_t)row * KD_DIM + j] = (bf16_t)(q * 0.125f);  // fold 1/sqrt(64)
    Kb[(size_t)row * KD_DIM + j] = (bf16_t)k;
}

// ---------------------------------------------------------------------------
// Kernel B: fused scores (bf16 WMMA) + streaming per-row top-8 + 8-elem softmax.
// 1 block = 16 rows, 512 threads = 16 waves. 16 chunks of 1024 columns.
// ---------------------------------------------------------------------------
__global__ __launch_bounds__(512) void scores_topk_kernel(
    const bf16_t* __restrict__ Qb, const bf16_t* __restrict__ Kb,
    float* __restrict__ out_vals, int* __restrict__ out_idx)
{
    __shared__ float sc[ROWS_PER_BLK * CHUNK];   // 64 KB score chunk

    const int tid     = threadIdx.x;
    const int lane    = tid & 31;
    const int wave    = tid >> 5;                // 0..15
    const int rowBase = blockIdx.x * ROWS_PER_BLK;
    const int l16     = lane & 15;
    const int hi      = lane >> 4;               // 0 | 1 (half-wave)
    const int e0      = hi ? 8 : 0;              // bf16 K-offset split per ISA layout

    // ---- A tile (16 Q rows) resident in registers for the whole kernel ----
    // 16-bit A layout: lane M=l16; lanes 0-15 hold K 0..7 & 16..23 in V0..7,
    // lanes 16-31 hold K 8..15 & 24..31 (ISA 7.12.2).
    const bf16_t* qrow = Qb + (size_t)(rowBase + l16) * KD_DIM;
    v8bf a00 = *(const v8bf*)(qrow + e0);
    v8bf a01 = *(const v8bf*)(qrow + e0 + 16);
    v8bf a10 = *(const v8bf*)(qrow + e0 + 32);
    v8bf a11 = *(const v8bf*)(qrow + e0 + 48);
    const v16bf A0 = __builtin_shufflevector(a00, a01, 0,1,2,3,4,5,6,7,8,9,10,11,12,13,14,15);
    const v16bf A1 = __builtin_shufflevector(a10, a11, 0,1,2,3,4,5,6,7,8,9,10,11,12,13,14,15);

    // ---- lane-local top-8 state (compile-time indexed -> stays in VGPRs) ----
    float tv[TOPK]; int ti[TOPK];
    #pragma unroll
    for (int i = 0; i < TOPK; ++i) { tv[i] = -3.4e38f; ti[i] = 0; }
    float tmin = -3.4e38f; int minslot = 0;

    for (int chunk = 0; chunk < N_ROWS / CHUNK; ++chunk) {
        // ================= score phase: 4 tiles per wave =================
        #pragma unroll
        for (int tt = 0; tt < (CHUNK / 16) / 16; ++tt) {
            const int tloc    = wave * 4 + tt;               // tile 0..63 in chunk
            const int colBase = chunk * CHUNK + tloc * 16;

            // B tile = K^T: lane N=l16, same K split as A (ISA B layout).
            const bf16_t* krow = Kb + (size_t)(colBase + l16) * KD_DIM;
            v8bf b00 = *(const v8bf*)(krow + e0);
            v8bf b01 = *(const v8bf*)(krow + e0 + 16);
            v8bf b10 = *(const v8bf*)(krow + e0 + 32);
            v8bf b11 = *(const v8bf*)(krow + e0 + 48);
            const v16bf B0 = __builtin_shufflevector(b00, b01, 0,1,2,3,4,5,6,7,8,9,10,11,12,13,14,15);
            const v16bf B1 = __builtin_shufflevector(b10, b11, 0,1,2,3,4,5,6,7,8,9,10,11,12,13,14,15);

            v8f c = {};
            c = __builtin_amdgcn_wmma_f32_16x16x32_bf16(false, A0, false, B0,
                                                        (short)0, c, false, false);
            c = __builtin_amdgcn_wmma_f32_16x16x32_bf16(false, A1, false, B1,
                                                        (short)0, c, false, false);

            // D layout: VGPR r, lanes 0-15 -> (M=r, N=lane); lanes 16-31 -> (M=r+8).
            const int colL = tloc * 16 + l16;
            #pragma unroll
            for (int r = 0; r < 8; ++r)
                sc[(r + hi * 8) * CHUNK + colL] = c[r];
        }
        __syncthreads();

        // ================= top-k phase: wave w scans row w =================
        const float* rowp  = sc + wave * CHUNK;
        const int    gBase = chunk * CHUNK;
        for (int cc = lane; cc < CHUNK; cc += 32) {
            const float v = rowp[cc];
            if (v > tmin) {
                const int g = gBase + cc;
                #pragma unroll
                for (int i = 0; i < TOPK; ++i)
                    if (i == minslot) { tv[i] = v; ti[i] = g; }
                tmin = tv[0]; minslot = 0;
                #pragma unroll
                for (int i = 1; i < TOPK; ++i)
                    if (tv[i] < tmin) { tmin = tv[i]; minslot = i; }
            }
        }
        __syncthreads();
    }

    // ====== merge 32 lane-local top-8 lists -> row top-8 (descending) ======
    float ordv[TOPK]; int ordi[TOPK];
    #pragma unroll
    for (int r = 0; r < TOPK; ++r) {
        float lmax = tv[0]; int lslot = 0;
        #pragma unroll
        for (int i = 1; i < TOPK; ++i)
            if (tv[i] > lmax) { lmax = tv[i]; lslot = i; }

        float m = lmax;
        #pragma unroll
        for (int off = 16; off > 0; off >>= 1)
            m = fmaxf(m, __shfl_xor(m, off, 32));

        const unsigned long long ball = __ballot(lmax == m);
        const int winner = (int)__builtin_ctzll(ball);

        int myidx = 0;
        #pragma unroll
        for (int i = 0; i < TOPK; ++i)
            if (i == lslot) myidx = ti[i];
        const int widx = __shfl(myidx, winner, 32);

        if (lane == winner) {
            #pragma unroll
            for (int i = 0; i < TOPK; ++i)
                if (i == lslot) tv[i] = -3.4e38f;   // remove extracted entry
        }
        ordv[r] = m; ordi[r] = widx;
    }

    // ====== 8-element softmax (denominator of full softmax cancels) ======
    if (lane == 0) {
        const int grow = rowBase + wave;
        const float m0 = ordv[0];
        float e[TOPK], sum = 0.f;
        #pragma unroll
        for (int r = 0; r < TOPK; ++r) { e[r] = __expf(ordv[r] - m0); sum += e[r]; }
        const float inv = 1.0f / sum;
        #pragma unroll
        for (int r = 0; r < TOPK; ++r) {
            out_vals[grow * TOPK + r] = e[r] * inv;
            out_idx [grow * TOPK + r] = ordi[r];
        }
    }
}

// ---------------------------------------------------------------------------
extern "C" void kernel_launch(void* const* d_in, const int* in_sizes, int n_in,
                              void* d_out, int out_size, void* d_ws, size_t ws_size,
                              hipStream_t stream)
{
    const float* U  = (const float*)d_in[0];   // [N, T, D]
    const float* Wq = (const float*)d_in[1];   // [D, KD]
    const float* bq = (const float*)d_in[2];   // [KD]
    const float* Wk = (const float*)d_in[3];   // [D, KD]
    const float* bk = (const float*)d_in[4];   // [KD]
    // d_in[5] = topk scalar (compile-time TOPK=8)

    bf16_t* Qb = (bf16_t*)d_ws;                         // 2 MB
    bf16_t* Kb = Qb + (size_t)N_ROWS * KD_DIM;          // 2 MB

    float* out_vals = (float*)d_out;                    // [N, 8] f32
    int*   out_idx  = (int*)d_out + (size_t)N_ROWS * TOPK;  // [N, 8] i32

    hipLaunchKernelGGL(pool_project_kernel, dim3(N_ROWS), dim3(64), 0, stream,
                       U, Wq, bq, Wk, bk, Qb, Kb);
    hipLaunchKernelGGL(scores_topk_kernel, dim3(N_ROWS / ROWS_PER_BLK), dim3(512),
                       0, stream, Qb, Kb, out_vals, out_idx);
}
